// SwinGNN_36077725287098
// MI455X (gfx1250) — matrix operand order repaired and checked by reading, not compiled, not run on hardware
//
#include <hip/hip_runtime.h>
#include <stdint.h>

typedef __bf16 v4bf  __attribute__((ext_vector_type(4)));
typedef __bf16 v8bf  __attribute__((ext_vector_type(8)));
typedef __bf16 v16bf __attribute__((ext_vector_type(16)));
typedef float  v4f   __attribute__((ext_vector_type(4)));
typedef float  v8f   __attribute__((ext_vector_type(8)));

// build a 16-wide bf16 fragment from two 16-byte chunks
__device__ __forceinline__ v16bf ld16x2(const __bf16* p0, const __bf16* p1) {
    v8bf lo = *(const v8bf*)p0;
    v8bf hi = *(const v8bf*)p1;
    return __builtin_shufflevector(lo, hi, 0,1,2,3,4,5,6,7,8,9,10,11,12,13,14,15);
}

__device__ __forceinline__ v8f wmma_bf16(v16bf a, v16bf b, v8f c) {
    return __builtin_amdgcn_wmma_f32_16x16x32_bf16(false, a, false, b, (short)0, c, false, false);
}

// ---------- prep: transpose + bf16-convert weights into workspace ----------
// W1t: [256][768] bf16 (row = out-channel, contiguous K)
// W2t: [128][256] bf16
__global__ void gcn_prep(const float* __restrict__ W1, const float* __restrict__ W2,
                         __bf16* __restrict__ W1t, __bf16* __restrict__ W2t) {
    int idx = blockIdx.x * 256 + threadIdx.x;
    if (idx < 256 * 768) {
        int n = idx / 768, k = idx % 768;
        W1t[idx] = (__bf16)W1[k * 256 + n];
    } else if (idx < 256 * 768 + 128 * 256) {
        int j = idx - 256 * 768;
        int n = j / 256, k = j % 256;
        W2t[j] = (__bf16)W2[k * 128 + n];
    }
}

// ---------- main: one workgroup per graph ----------
// LDS layout (bytes):
//   As : [0, 9216)                 64x72  bf16   normalized adjacency, padded
//   Xs : [9216, 108544)            64x776 bf16   staged X (aliased later by H1, pooled)
//   T1 : [108544, 145408)          256x72 bf16   transposed intermediates (aliased by T2)
#define LDS_BYTES (9216 + 99328 + 36864)

__global__ __launch_bounds__(256)
void gcn_main(const float* __restrict__ x,
              const float* __restrict__ b1, const float* __restrict__ b2,
              const float* __restrict__ Wf, const float* __restrict__ bfv,
              const __bf16* __restrict__ W1t,
              const __bf16* __restrict__ W2t,
              float* __restrict__ out) {
    extern __shared__ char smem[];
    __bf16* As = (__bf16*)smem;                      // 64 x 72
    __bf16* Xs = (__bf16*)(smem + 9216);             // 64 x 776
    __bf16* T1 = (__bf16*)(smem + 9216 + 99328);     // 256 x 72
    __bf16* H1 = Xs;                                 // 64 x 264 (alias, after GEMM1)
    __bf16* T2 = T1;                                 // 128 x 72 (alias, after phase 3)
    float* pooled = (float*)Xs;                      // 128 f32 (alias, after phase 4)

    const int tid  = threadIdx.x;
    const int w    = tid >> 5;        // wave 0..7
    const int lane = tid & 31;
    const int half = lane >> 4;       // 0 or 1
    const int l16  = lane & 15;

    // ---- phase 0: synthesize normalized padded adjacency (64x64, bf16) ----
    for (int idx = tid; idx < 64 * 64; idx += 256) {
        int i = idx >> 6, j = idx & 63;
        float v = 0.f;
        if (i < 49 && j < 49) {
            int ri = i / 7, ci = i % 7, rj = j / 7, cj = j % 7;
            int dr = ri - rj, dc = ci - cj;
            if (dr >= -1 && dr <= 1 && dc >= -1 && dc <= 1) {
                float di = (float)(((ri > 0) + (ri < 6) + 1) * ((ci > 0) + (ci < 6) + 1));
                float dj = (float)(((rj > 0) + (rj < 6) + 1) * ((cj > 0) + (cj < 6) + 1));
                v = rsqrtf(di) * rsqrtf(dj);
            }
        }
        As[i * 72 + j] = (__bf16)v;
    }

    // ---- phase 1: stage X_b as bf16 into LDS [64][776], zero pad rows 49..63 ----
    const v4f* xb = (const v4f*)(x + (size_t)blockIdx.x * 49 * 768);
    for (int i4 = tid; i4 < 49 * 192; i4 += 256) {
        int row = i4 / 192, c4 = i4 % 192;
        v4f f = xb[i4];
        *(v4bf*)(&Xs[row * 776 + c4 * 4]) = __builtin_convertvector(f, v4bf);
    }
    {
        unsigned* Xsu = (unsigned*)(Xs + 49 * 776);
        for (int idx = tid; idx < 15 * 388; idx += 256) Xsu[idx] = 0;
    }
    __syncthreads();

    const v8f vz = {0.f, 0.f, 0.f, 0.f, 0.f, 0.f, 0.f, 0.f};
    const int nt0 = w * 2;   // waves split the 16 N-tiles of the 256-wide GEMMs

    // ---- phase 2: GEMM1  XW1 = Xs[64x768] @ W1[768x256] ----
    {
        v8f acc[4][2];
        #pragma unroll
        for (int mt = 0; mt < 4; ++mt) { acc[mt][0] = vz; acc[mt][1] = vz; }
        for (int kt = 0; kt < 24; ++kt) {
            const int k0 = kt * 32;
            const int koff = k0 + half * 8;
            v16bf afr[4];
            #pragma unroll
            for (int mt = 0; mt < 4; ++mt) {
                const __bf16* pr = Xs + (mt * 16 + l16) * 776 + koff;
                afr[mt] = ld16x2(pr, pr + 16);
            }
            v16bf bfr[2];
            #pragma unroll
            for (int j = 0; j < 2; ++j) {
                int n = (nt0 + j) * 16 + l16;
                bfr[j] = ld16x2(W1t + n * 768 + koff, W1t + n * 768 + koff + 16);
            }
            #pragma unroll
            for (int mt = 0; mt < 4; ++mt)
                #pragma unroll
                for (int j = 0; j < 2; ++j)
                    acc[mt][j] = wmma_bf16(afr[mt], bfr[j], acc[mt][j]);
        }
        // store transposed: T1[col][row] — packed 16B LDS stores
        #pragma unroll
        for (int mt = 0; mt < 4; ++mt)
            #pragma unroll
            for (int j = 0; j < 2; ++j) {
                int n = (nt0 + j) * 16 + l16;
                *(v8bf*)(&T1[n * 72 + mt * 16 + half * 8]) =
                    __builtin_convertvector(acc[mt][j], v8bf);
            }
    }
    __syncthreads();

    // ---- phase 3: H1 = relu(A @ XW1 + b1), row-major bf16 [64][264] ----
    {
        v8f acc[4][2];
        #pragma unroll
        for (int mt = 0; mt < 4; ++mt) { acc[mt][0] = vz; acc[mt][1] = vz; }
        #pragma unroll
        for (int ks = 0; ks < 2; ++ks) {
            const int k0 = ks * 32;
            const int koff = k0 + half * 8;
            v16bf afr[4];
            #pragma unroll
            for (int mt = 0; mt < 4; ++mt) {
                const __bf16* pr = As + (mt * 16 + l16) * 72 + koff;
                afr[mt] = ld16x2(pr, pr + 16);
            }
            v16bf bfr[2];
            #pragma unroll
            for (int j = 0; j < 2; ++j) {
                int n = (nt0 + j) * 16 + l16;
                const __bf16* pb = T1 + n * 72 + koff;
                bfr[j] = ld16x2(pb, pb + 16);
            }
            #pragma unroll
            for (int mt = 0; mt < 4; ++mt)
                #pragma unroll
                for (int j = 0; j < 2; ++j)
                    acc[mt][j] = wmma_bf16(afr[mt], bfr[j], acc[mt][j]);
        }
        float bb0 = b1[nt0 * 16 + l16];
        float bb1 = b1[(nt0 + 1) * 16 + l16];
        #pragma unroll
        for (int mt = 0; mt < 4; ++mt)
            #pragma unroll
            for (int j = 0; j < 2; ++j) {
                int n = (nt0 + j) * 16 + l16;
                float bb = j ? bb1 : bb0;
                #pragma unroll
                for (int v = 0; v < 8; ++v) {
                    int m = mt * 16 + half * 8 + v;
                    H1[m * 264 + n] = (__bf16)fmaxf(acc[mt][j][v] + bb, 0.f);
                }
            }
    }
    __syncthreads();

    // ---- phase 4: H1W2 = H1[64x256] @ W2[256x128], store transposed T2[128][72] ----
    {
        v8f acc[4];
        #pragma unroll
        for (int mt = 0; mt < 4; ++mt) acc[mt] = vz;
        const int n = w * 16 + l16;           // out channel 0..127 (wave = N-tile)
        for (int kt = 0; kt < 8; ++kt) {
            const int k0 = kt * 32;
            const int koff = k0 + half * 8;
            v16bf afr[4];
            #pragma unroll
            for (int mt = 0; mt < 4; ++mt) {
                const __bf16* pr = H1 + (mt * 16 + l16) * 264 + koff;
                afr[mt] = ld16x2(pr, pr + 16);
            }
            v16bf bfr = ld16x2(W2t + n * 256 + koff, W2t + n * 256 + koff + 16);
            #pragma unroll
            for (int mt = 0; mt < 4; ++mt) acc[mt] = wmma_bf16(afr[mt], bfr, acc[mt]);
        }
        #pragma unroll
        for (int mt = 0; mt < 4; ++mt)
            *(v8bf*)(&T2[n * 72 + mt * 16 + half * 8]) =
                __builtin_convertvector(acc[mt], v8bf);
    }
    __syncthreads();

    // ---- phase 5: H2 = relu(A @ H1W2 + b2); masked mean-pool in registers ----
    {
        v8f acc[4];
        #pragma unroll
        for (int mt = 0; mt < 4; ++mt) acc[mt] = vz;
        const int n = w * 16 + l16;
        #pragma unroll
        for (int ks = 0; ks < 2; ++ks) {
            const int k0 = ks * 32;
            const int koff = k0 + half * 8;
            v16bf afr[4];
            #pragma unroll
            for (int mt = 0; mt < 4; ++mt) {
                const __bf16* pr = As + (mt * 16 + l16) * 72 + koff;
                afr[mt] = ld16x2(pr, pr + 16);
            }
            const __bf16* pb = T2 + n * 72 + koff;
            v16bf bfr = ld16x2(pb, pb + 16);
            #pragma unroll
            for (int mt = 0; mt < 4; ++mt) acc[mt] = wmma_bf16(afr[mt], bfr, acc[mt]);
        }
        float b2v = b2[n];
        float s = 0.f;
        #pragma unroll
        for (int mt = 0; mt < 4; ++mt)
            #pragma unroll
            for (int v = 0; v < 8; ++v) {
                int m = mt * 16 + half * 8 + v;
                float h = fmaxf(acc[mt][v] + b2v, 0.f);
                if (m < 49) s += h;   // exclude zero-padded rows
            }
        s += __shfl_xor(s, 16);       // combine half-lane row groups
        if (lane < 16) pooled[w * 16 + lane] = s * (1.f / 49.f);
    }
    __syncthreads();

    // ---- phase 6: out = pooled @ Wf + bf  (13 outputs) ----
    if (tid < 13) {
        float a = bfv[tid];
        #pragma unroll 4
        for (int p = 0; p < 128; ++p) a += pooled[p] * Wf[p * 13 + tid];
        out[(size_t)blockIdx.x * 13 + tid] = a;
    }
}

extern "C" void kernel_launch(void* const* d_in, const int* in_sizes, int n_in,
                              void* d_out, int out_size, void* d_ws, size_t ws_size,
                              hipStream_t stream) {
    const float* x   = (const float*)d_in[0];
    const float* W1  = (const float*)d_in[1];
    const float* b1  = (const float*)d_in[2];
    const float* W2  = (const float*)d_in[3];
    const float* b2  = (const float*)d_in[4];
    const float* Wf  = (const float*)d_in[5];
    const float* bf_ = (const float*)d_in[6];
    float* out = (float*)d_out;

    __bf16* W1t = (__bf16*)d_ws;              // 256*768 bf16
    __bf16* W2t = W1t + 256 * 768;            // 128*256 bf16

    // allow >64KB dynamic LDS (gfx1250 WGP has 320KB)
    hipFuncSetAttribute((const void*)gcn_main,
                        hipFuncAttributeMaxDynamicSharedMemorySize, LDS_BYTES);

    const int prepElems = 256 * 768 + 128 * 256;
    gcn_prep<<<(prepElems + 255) / 256, 256, 0, stream>>>(W1, W2, W1t, W2t);
    gcn_main<<<4096, 256, LDS_BYTES, stream>>>(x, b1, b2, Wf, bf_, W1t, W2t, out);
}
